// MatchModel_21062519619910
// MI455X (gfx1250) — matrix operand order repaired and checked by reading, not compiled, not run on hardware
//
#include <hip/hip_runtime.h>
#include <hip/hip_bf16.h>

typedef __attribute__((ext_vector_type(16))) _Float16 v16h;
typedef __attribute__((ext_vector_type(8)))  float    v8f;
typedef __attribute__((ext_vector_type(4)))  float    v4f;

#define HEADS 2
#define DHEAD 64
#define HD    128
#define NEG_SLOPE 0.2f

// ---------- order-preserving float <-> uint key (for atomic max) ----------
__device__ __forceinline__ unsigned fenc(float f) {
  unsigned u = __float_as_uint(f);
  return (u & 0x80000000u) ? ~u : (u | 0x80000000u);
}
__device__ __forceinline__ float fdec(unsigned k) {
  if (k == 0u) return 0.0f;  // no edges hit this dst -> reference maps to 0
  return (k & 0x80000000u) ? __uint_as_float(k & 0x7fffffffu)
                           : __uint_as_float(~k);
}

// ---------- WMMA matmul: D[M,N] = gather(A)[M,K] @ B[K,N] (+bias)(relu) ----------
// One 16 x N row-strip per wave: the A fragment (big HBM operand) is loaded once
// per K-step and reused across all NT column tiles (NT accumulators in VGPRs).
// B is staged in LDS as f16 PRE-SWIZZLED into WMMA fragment order; all NT
// B fragments of a K-step are preloaded into registers so the ds_load latency
// overlaps with WMMA issue instead of stalling at s_wait_dscnt 0 per fragment.
template <int K, int NT, bool BIAS, bool RELU>
__global__ __launch_bounds__(128) void mm_wmma(
    const float* __restrict__ A, const int* __restrict__ rowidx,
    const float* __restrict__ B, float* __restrict__ D,
    int Mtiles, const float* __restrict__ bias)
{
  constexpr int N = NT * 16;
  constexpr int KT = K / 32;
  __shared__ _Float16 sB[KT * NT * 32 * 16];

  // Stage B (K x N, f32 row-major) -> LDS f16 in fragment order:
  // sB[((kkI*NT + nt)*32 + lane)*16 + i] = B[(kkI*32 + (i>>3)*16 + (lane>>4)*8 + (i&7))][nt*16 + (lane&15)]
  for (int idx = threadIdx.x; idx < K * N; idx += 128) {
    int i    = idx & 15;
    int l    = (idx >> 4) & 31;
    int tile = idx >> 9;
    int nt   = tile % NT;
    int kkI  = tile / NT;
    int k    = kkI * 32 + (i >> 3) * 16 + (l >> 4) * 8 + (i & 7);
    int col  = nt * 16 + (l & 15);
    sB[idx] = (_Float16)B[k * N + col];
  }
  __syncthreads();

  const int wave = threadIdx.x >> 5;
  const int lane = threadIdx.x & 31;
  const int tm   = blockIdx.x * 4 + wave;
  if (tm >= Mtiles) return;               // wave-uniform; EXEC stays all-ones

  const int hi  = lane >> 4;
  const int c16 = lane & 15;
  int arow = tm * 16 + c16;               // A row for this lane
  if (rowidx) arow = rowidx[arow];
  const float* __restrict__ Ar = A + (long)arow * K;

  v8f zero = {};
  v8f acc[NT];
#pragma unroll
  for (int nt = 0; nt < NT; ++nt) acc[nt] = zero;

#pragma unroll
  for (int kkI = 0; kkI < KT; ++kkI) {
    // A fragment: lanes<16 hold K = kk+{0..7,16..23}; lanes>=16 K = kk+{8..15,24..31}
    const int kb = kkI * 32 + hi * 8;
    v4f f0 = *(const v4f*)(Ar + kb);
    v4f f1 = *(const v4f*)(Ar + kb + 4);
    v4f f2 = *(const v4f*)(Ar + kb + 16);
    v4f f3 = *(const v4f*)(Ar + kb + 20);
    v16h a;
    a[0]=(_Float16)f0.x; a[1]=(_Float16)f0.y; a[2]=(_Float16)f0.z; a[3]=(_Float16)f0.w;
    a[4]=(_Float16)f1.x; a[5]=(_Float16)f1.y; a[6]=(_Float16)f1.z; a[7]=(_Float16)f1.w;
    a[8]=(_Float16)f2.x; a[9]=(_Float16)f2.y; a[10]=(_Float16)f2.z; a[11]=(_Float16)f2.w;
    a[12]=(_Float16)f3.x; a[13]=(_Float16)f3.y; a[14]=(_Float16)f3.z; a[15]=(_Float16)f3.w;

    // Preload all NT B fragments for this K-step (back-to-back ds_load_b128),
    // then issue the NT WMMAs; waits become staggered instead of dscnt==0 stalls.
    v16h b[NT];
#pragma unroll
    for (int nt = 0; nt < NT; ++nt)
      b[nt] = *(const v16h*)(sB + ((kkI * NT + nt) * 32 + lane) * 16);
#pragma unroll
    for (int nt = 0; nt < NT; ++nt)
      acc[nt] = __builtin_amdgcn_wmma_f32_16x16x32_f16(
          false, a, false, b[nt], (short)0, acc[nt], false, false);
  }

#pragma unroll
  for (int nt = 0; nt < NT; ++nt) {
    const int colg = nt * 16 + c16;
    const float bv = BIAS ? bias[colg] : 0.0f;
#pragma unroll
    for (int j = 0; j < 8; ++j) {         // D: VGPR j -> row j (lanes<16) / j+8 (lanes>=16)
      const int row = tm * 16 + hi * 8 + j;
      float v = acc[nt][j] + bv;
      if (RELU) v = fmaxf(v, 0.0f);
      D[(long)row * N + colg] = v;
    }
  }
}

// ---------- attention scores: out[n,h] = sum_d feat[n,h,d]*avec[h,d] ----------
__global__ void attn_score(const float* __restrict__ feat,
                           const float* __restrict__ avec,
                           float* __restrict__ out, int n /* nodes*HEADS */)
{
  int g = blockIdx.x * blockDim.x + threadIdx.x;
  if (g >= n) return;
  int h = g & (HEADS - 1);
  const float* f = feat + (long)(g >> 1) * HD + h * DHEAD;
  const float* a = avec + h * DHEAD;
  float s = 0.f;
#pragma unroll 8
  for (int d = 0; d < DHEAD; ++d) s += f[d] * a[d];
  out[g] = s;
}

__global__ void zero_u32(unsigned* __restrict__ p, int n) {
  int g = blockIdx.x * blockDim.x + threadIdx.x;
  if (g < n) p[g] = 0u;
}

__global__ void init_bias(float* __restrict__ out, const float* __restrict__ bias,
                          long total) {
  long g = (long)blockIdx.x * blockDim.x + threadIdx.x;
  if (g < total) out[g] = bias[g & (HD - 1)];
}

// per-edge: e = leaky_relu(el[src]+er[dst]); stash e; atomic segment max (uint key)
__global__ void edge_max(const int* __restrict__ src, const int* __restrict__ dst,
                         const float* __restrict__ el, const float* __restrict__ er,
                         float* __restrict__ exbuf, unsigned* __restrict__ mkey, int nE)
{
  int e = blockIdx.x * blockDim.x + threadIdx.x;
  if (e >= nE) return;
  int s = src[e], d = dst[e];
#pragma unroll
  for (int h = 0; h < HEADS; ++h) {
    float v = el[s * HEADS + h] + er[d * HEADS + h];
    v = v > 0.f ? v : NEG_SLOPE * v;
    exbuf[e * HEADS + h] = v;
    atomicMax(&mkey[d * HEADS + h], fenc(v));
  }
}

// per-edge: ex = exp(e - m[dst]); stash ex; atomic segment sum of denominators
__global__ void edge_expsum(const int* __restrict__ dst, float* __restrict__ exbuf,
                            const unsigned* __restrict__ mkey, float* __restrict__ den,
                            int nE)
{
  int e = blockIdx.x * blockDim.x + threadIdx.x;
  if (e >= nE) return;
  int d = dst[e];
#pragma unroll
  for (int h = 0; h < HEADS; ++h) {
    float m  = fdec(mkey[d * HEADS + h]);
    float ex = __expf(exbuf[e * HEADS + h] - m);
    exbuf[e * HEADS + h] = ex;
    atomicAdd(&den[d * HEADS + h], ex);
  }
}

// per-(edge, feature): out[dst, j] += alpha * hs[src, j]   (out pre-init'd with bias)
__global__ void edge_scatter(const int* __restrict__ src, const int* __restrict__ dst,
                             const float* __restrict__ hs, const float* __restrict__ exbuf,
                             const float* __restrict__ den, float* __restrict__ out,
                             long total /* nE * HD */)
{
  long g = (long)blockIdx.x * blockDim.x + threadIdx.x;
  if (g >= total) return;
  int  j = (int)(g & (HD - 1));
  long e = g >> 7;
  int  h = j >> 6;
  int  s = src[e], d = dst[e];
  float alpha = exbuf[e * HEADS + h] / fmaxf(den[d * HEADS + h], 1e-9f);
  atomicAdd(&out[(long)d * HD + j], alpha * hs[(long)s * HD + j]);
}

// cat = [u_n (256x128) | q_emb (256x64)] -> 256x192
__global__ void concat_cat(const float* __restrict__ un, const float* __restrict__ q,
                           float* __restrict__ cat, int total)
{
  int g = blockIdx.x * blockDim.x + threadIdx.x;
  if (g >= total) return;
  int r = g / 192, c = g - r * 192;
  cat[g] = (c < 128) ? un[r * 128 + c] : q[r * 64 + (c - 128)];
}

// ============================ host side ============================
namespace {

struct Ptrs {
  const float *user_emb, *uq_emb, *item_emb, *iq_emb, *map_W, *map_b;
  // per-GAT: Wsrc, Wdst, al, ar, b
  const float *gW[5][5];
  const int *su_src, *su_dst, *si_src, *si_dst, *ck_src, *ck_dst, *batch;
};

static void run_mm(const float* A, const int* rowidx, const float* B, float* D,
                   int M, int N, int K, const float* bias, int relu, hipStream_t s)
{
  int Mtiles = M / 16;
  int blocks = (Mtiles + 3) / 4;
  if (N == 128 && K == 64)
    mm_wmma<64, 8, false, false><<<blocks, 128, 0, s>>>(A, rowidx, B, D, Mtiles, bias);
  else if (N == 128 && K == 128)
    mm_wmma<128, 8, false, false><<<blocks, 128, 0, s>>>(A, rowidx, B, D, Mtiles, bias);
  else if (N == 128 && K == 192)
    mm_wmma<192, 8, false, false><<<blocks, 128, 0, s>>>(A, rowidx, B, D, Mtiles, bias);
  else /* N == 64 && K == 128, final bias+relu */
    mm_wmma<128, 4, true, true><<<blocks, 128, 0, s>>>(A, rowidx, B, D, Mtiles, bias);
  (void)relu;
}

static void run_gat_edges(const int* src, const int* dst, int nE, int nDst,
                          const float* el, const float* er, float* exbuf,
                          unsigned* mkey, float* den, const float* hs,
                          const float* bias, float* out, hipStream_t s)
{
  int n2 = nDst * HEADS;
  zero_u32<<<(n2 + 255) / 256, 256, 0, s>>>(mkey, n2);
  zero_u32<<<(n2 + 255) / 256, 256, 0, s>>>((unsigned*)den, n2);
  edge_max<<<(nE + 255) / 256, 256, 0, s>>>(src, dst, el, er, exbuf, mkey, nE);
  edge_expsum<<<(nE + 255) / 256, 256, 0, s>>>(dst, exbuf, mkey, den, nE);
  long tb = (long)nDst * HD;
  init_bias<<<(unsigned)((tb + 255) / 256), 256, 0, s>>>(out, bias, tb);
  long ts = (long)nE * HD;
  edge_scatter<<<(unsigned)((ts + 127) / 128), 128, 0, s>>>(src, dst, hs, exbuf,
                                                            den, out, ts);
}

} // namespace

extern "C" void kernel_launch(void* const* d_in, const int* in_sizes, int n_in,
                              void* d_out, int out_size, void* d_ws, size_t ws_size,
                              hipStream_t stream)
{
  Ptrs P;
  if (n_in >= 38 && in_sizes[0] == 6400000) {
    // setup_inputs() insertion order: params leaves then edge arrays
    P.user_emb = (const float*)d_in[0];  P.uq_emb = (const float*)d_in[1];
    P.item_emb = (const float*)d_in[2];  P.iq_emb = (const float*)d_in[3];
    P.map_W    = (const float*)d_in[4];  P.map_b  = (const float*)d_in[5];
    for (int g = 0; g < 5; ++g)
      for (int k = 0; k < 5; ++k)
        P.gW[g][k] = (const float*)d_in[6 + g * 5 + k];  // Wsrc,Wdst,al,ar,b
    P.su_src = (const int*)d_in[31]; P.su_dst = (const int*)d_in[32];
    P.si_src = (const int*)d_in[33]; P.si_dst = (const int*)d_in[34];
    P.ck_src = (const int*)d_in[35]; P.ck_dst = (const int*)d_in[36];
    P.batch  = (const int*)d_in[37];
  } else {
    // jax tree-flatten order (keys sorted at each level; 'W' < lowercase)
    P.batch  = (const int*)d_in[0];
    P.ck_dst = (const int*)d_in[1];
    P.ck_src = (const int*)d_in[2];
    auto fill = [&](int g, int base) {
      P.gW[g][1] = (const float*)d_in[base + 0];  // Wdst
      P.gW[g][0] = (const float*)d_in[base + 1];  // Wsrc
      P.gW[g][2] = (const float*)d_in[base + 2];  // al
      P.gW[g][3] = (const float*)d_in[base + 3];  // ar
      P.gW[g][4] = (const float*)d_in[base + 4];  // b
    };
    fill(4, 3);   // clicked
    fill(3, 8);   // denotes_i
    fill(1, 13);  // denotes_u
    P.item_emb = (const float*)d_in[18];
    P.iq_emb   = (const float*)d_in[19];
    P.map_W    = (const float*)d_in[20];
    P.map_b    = (const float*)d_in[21];
    fill(2, 22);  // shares_i
    fill(0, 27);  // shares_u
    P.si_dst = (const int*)d_in[32]; P.si_src = (const int*)d_in[33];
    P.su_dst = (const int*)d_in[34]; P.su_src = (const int*)d_in[35];
    P.uq_emb   = (const float*)d_in[36];
    P.user_emb = (const float*)d_in[37];
  }

  // ---- workspace bump allocator (reuse big buffers across phases) ----
  char* w = (char*)d_ws;
  auto alloc = [&](size_t bytes) -> void* {
    void* p = (void*)w;
    w += (bytes + 255) & ~(size_t)255;
    return p;
  };
  float*    hsB   = (float*)alloc(100000ull * HD * 4);  // hs1 / hs3 / hs5
  float*    hdB   = (float*)alloc(100000ull * HD * 4);  // hd2 / hd4 / hd5
  float*    u_h   = (float*)alloc(100000ull * HD * 4);
  float*    i_h   = (float*)alloc( 50000ull * HD * 4);
  float*    h5    = (float*)alloc(100000ull * HD * 4);
  float*    smD   = (float*)alloc(256ull * HD * 4);     // hd1 / hd3 (code side)
  float*    smN   = (float*)alloc(256ull * HD * 4);     // u_n / i_n
  float*    catB  = (float*)alloc(256ull * 192 * 4);
  float*    smS   = (float*)alloc(256ull * HD * 4);     // hs2 / hs4
  float*    elB   = (float*)alloc(100000ull * HEADS * 4);
  float*    erB   = (float*)alloc(100000ull * HEADS * 4);
  unsigned* mkey  = (unsigned*)alloc(100000ull * HEADS * 4);
  float*    den   = (float*)alloc(100000ull * HEADS * 4);
  float*    exB   = (float*)alloc(500000ull * HEADS * 4);
  (void)ws_size; (void)out_size; (void)n_in;

  const int NU = 100000, NI = 50000, NC = 256;

  // ================= users: user -> code (shares_u) =================
  run_mm(P.user_emb, nullptr, P.gW[0][0], hsB, NU, 128, 64, nullptr, 0, stream);
  run_mm(P.uq_emb,   nullptr, P.gW[0][1], smD, NC, 128, 64, nullptr, 0, stream);
  attn_score<<<(NU * 2 + 255) / 256, 256, 0, stream>>>(hsB, P.gW[0][2], elB, NU * 2);
  attn_score<<<(NC * 2 + 255) / 256, 256, 0, stream>>>(smD, P.gW[0][3], erB, NC * 2);
  run_gat_edges(P.su_src, P.su_dst, NU, NC, elB, erB, exB, mkey, den,
                hsB, P.gW[0][4], smN /*u_n*/, stream);
  concat_cat<<<(NC * 192 + 255) / 256, 256, 0, stream>>>(smN, P.uq_emb, catB, NC * 192);

  // ================= users: code -> user (denotes_u) =================
  run_mm(catB,       nullptr, P.gW[1][0], smS, NC, 128, 192, nullptr, 0, stream);
  run_mm(P.user_emb, nullptr, P.gW[1][1], hdB, NU, 128,  64, nullptr, 0, stream);
  attn_score<<<(NC * 2 + 255) / 256, 256, 0, stream>>>(smS, P.gW[1][2], elB, NC * 2);
  attn_score<<<(NU * 2 + 255) / 256, 256, 0, stream>>>(hdB, P.gW[1][3], erB, NU * 2);
  run_gat_edges(P.su_dst, P.su_src, NU, NU, elB, erB, exB, mkey, den,
                smS, P.gW[1][4], u_h, stream);

  // ================= items: item -> code (shares_i) =================
  run_mm(P.item_emb, nullptr, P.gW[2][0], hsB, NI, 128, 64, nullptr, 0, stream);
  run_mm(P.iq_emb,   nullptr, P.gW[2][1], smD, NC, 128, 64, nullptr, 0, stream);
  attn_score<<<(NI * 2 + 255) / 256, 256, 0, stream>>>(hsB, P.gW[2][2], elB, NI * 2);
  attn_score<<<(NC * 2 + 255) / 256, 256, 0, stream>>>(smD, P.gW[2][3], erB, NC * 2);
  run_gat_edges(P.si_src, P.si_dst, NI, NC, elB, erB, exB, mkey, den,
                hsB, P.gW[2][4], smN /*i_n*/, stream);
  concat_cat<<<(NC * 192 + 255) / 256, 256, 0, stream>>>(smN, P.iq_emb, catB, NC * 192);

  // ================= items: code -> item (denotes_i) =================
  run_mm(catB,       nullptr, P.gW[3][0], smS, NC, 128, 192, nullptr, 0, stream);
  run_mm(P.item_emb, nullptr, P.gW[3][1], hdB, NI, 128,  64, nullptr, 0, stream);
  attn_score<<<(NC * 2 + 255) / 256, 256, 0, stream>>>(smS, P.gW[3][2], elB, NC * 2);
  attn_score<<<(NI * 2 + 255) / 256, 256, 0, stream>>>(hdB, P.gW[3][3], erB, NI * 2);
  run_gat_edges(P.si_dst, P.si_src, NI, NI, elB, erB, exB, mkey, den,
                smS, P.gW[3][4], i_h, stream);

  // ================= clicks: item -> user (clicked) =================
  run_mm(i_h, nullptr, P.gW[4][0], hsB, NI, 128, 128, nullptr, 0, stream);
  run_mm(u_h, nullptr, P.gW[4][1], hdB, NU, 128, 128, nullptr, 0, stream);
  attn_score<<<(NI * 2 + 255) / 256, 256, 0, stream>>>(hsB, P.gW[4][2], elB, NI * 2);
  attn_score<<<(NU * 2 + 255) / 256, 256, 0, stream>>>(hdB, P.gW[4][3], erB, NU * 2);
  run_gat_edges(P.ck_src, P.ck_dst, 500000, NU, elB, erB, exB, mkey, den,
                hsB, P.gW[4][4], h5, stream);

  // ============ final: relu(h[batch] @ map_W + map_b) -> d_out ============
  run_mm(h5, P.batch, P.map_W, (float*)d_out, 4096, 64, 128, P.map_b, 1, stream);
}
